// RoAttentionLayer_64914135712470
// MI455X (gfx1250) — compile-verified, matrix operand order checked by reading, not run on hardware
//
#include <hip/hip_runtime.h>
#include <hip/hip_bf16.h>

// ---------------------------------------------------------------------------
// RoPE multi-head attention for MI455X (gfx1250), wave32 + WMMA f16.
// B=4, H=8, L=2048, D=32, M=256.  Outputs: out [B,L,256] f32, series [B,H,L,L] f32.
// Fused logits+softmax+PV kernel keeps the 16x2048 logits block in LDS (128 KB);
// the 537 MB series tensor is written exactly once, streamed straight from LDS
// with GLOBAL_STORE_ASYNC_FROM_LDS_B128 (no VGPR staging, ASYNCcnt-tracked).
// ---------------------------------------------------------------------------

typedef __attribute__((ext_vector_type(8)))  _Float16 v8h;
typedef __attribute__((ext_vector_type(16))) _Float16 v16h;
typedef __attribute__((ext_vector_type(8)))  float    v8f;

#define B_DIM 4
#define H_DIM 8
#define L_DIM 2048
#define D_DIM 32
#define M_DIM 256

// Per-lane K pattern for 16-bit A(16x32) and B(32x16) is identical (ISA 7.12.2):
// lane<16 : K in {0..7} U {16..23};  lane>=16 : K in {8..15} U {24..31}.
__device__ __forceinline__ v16h ldtile_h(const _Float16* rowbase, int ko) {
  v8h lo = *reinterpret_cast<const v8h*>(rowbase + ko);
  v8h hi = *reinterpret_cast<const v8h*>(rowbase + ko + 16);
  return __builtin_shufflevector(lo, hi, 0,1,2,3,4,5,6,7,8,9,10,11,12,13,14,15);
}

__device__ __forceinline__ v16h ldtile_f(const float* rowbase, int ko) {
  v8f lo = *reinterpret_cast<const v8f*>(rowbase + ko);
  v8f hi = *reinterpret_cast<const v8f*>(rowbase + ko + 16);
  v16h r;
#pragma unroll
  for (int i = 0; i < 8; ++i) { r[i] = (_Float16)lo[i]; r[i + 8] = (_Float16)hi[i]; }
  return r;
}

__device__ __forceinline__ v8f wmma_f16(v16h a, v16h b, v8f c) {
  return __builtin_amdgcn_wmma_f32_16x16x32_f16(false, a, false, b, (short)0, c,
                                                false, false);
}

// --------------------------- pack kernels ----------------------------------

__global__ void pack_x_kernel(const float* __restrict__ x, _Float16* __restrict__ xh,
                              int n) {
  int i = blockIdx.x * blockDim.x + threadIdx.x;
  if (i < n) xh[i] = (_Float16)x[i];
}

// Wt[n][k] = W(k, n%256) for n in 0..767 (Wq|Wk|Wv), plus concatenated bias.
__global__ void pack_wqkv_kernel(const float* __restrict__ Wq, const float* __restrict__ Wk,
                                 const float* __restrict__ Wv, const float* __restrict__ bq,
                                 const float* __restrict__ bk, const float* __restrict__ bv,
                                 _Float16* __restrict__ Wt, float* __restrict__ bias_cat) {
  int idx = blockIdx.x * blockDim.x + threadIdx.x;
  if (idx >= 768 * 256) return;
  int n = idx >> 8;
  int k = idx & 255;
  const float* W = (n < 256) ? Wq : (n < 512) ? Wk : Wv;
  int nc = n & 255;
  Wt[(size_t)n * 256 + k] = (_Float16)W[(size_t)k * 256 + nc];
  if (k == 0) {
    const float* bb = (n < 256) ? bq : (n < 512) ? bk : bv;
    bias_cat[n] = bb[nc];
  }
}

__global__ void pack_wo_kernel(const float* __restrict__ Wo, _Float16* __restrict__ Wto) {
  int idx = blockIdx.x * blockDim.x + threadIdx.x;
  if (idx >= 256 * 256) return;
  int n = idx >> 8;
  int k = idx & 255;
  Wto[(size_t)n * 256 + k] = (_Float16)Wo[(size_t)k * 256 + n];
}

// --------------------------- generic WMMA GEMM -----------------------------
// C[M,N] = A[M,K](f16) * Bt[N,K](f16, pre-transposed) + bias[n]
// one wave computes one 16x16 tile of C; store f16 (Ch) or f32 (Cf).

__global__ void gemm_wmma_kernel(const _Float16* __restrict__ A,
                                 const _Float16* __restrict__ Bt,
                                 const float* __restrict__ bias,
                                 float* __restrict__ Cf, _Float16* __restrict__ Ch,
                                 int M, int N, int K) {
  int lane = threadIdx.x & 31;
  int gw = (blockIdx.x * blockDim.x + threadIdx.x) >> 5;
  int tiles_n = N >> 4;
  int tiles_m = M >> 4;
  int tm = gw / tiles_n;
  int tn = gw % tiles_n;
  if (tm >= tiles_m) return;  // wave-uniform

  int ko = (lane & 16) ? 8 : 0;
  const _Float16* arow = A + (size_t)(tm * 16 + (lane & 15)) * K;
  const _Float16* brow = Bt + (size_t)(tn * 16 + (lane & 15)) * K;

  v8f acc = {};
#pragma unroll 4
  for (int k = 0; k < K; k += 32) {
    v16h a = ldtile_h(arow + k, ko);
    v16h b = ldtile_h(brow + k, ko);
    acc = wmma_f16(a, b, acc);
  }

  int n = tn * 16 + (lane & 15);
  float bn = bias ? bias[n] : 0.0f;
  int m0 = tm * 16 + ((lane & 16) ? 8 : 0);
  if (Ch) {
#pragma unroll
    for (int i = 0; i < 8; ++i) Ch[(size_t)(m0 + i) * N + n] = (_Float16)(acc[i] + bn);
  } else {
#pragma unroll
    for (int i = 0; i < 8; ++i) Cf[(size_t)(m0 + i) * N + n] = acc[i] + bn;
  }
}

// --------------------------- RoPE + relayout -------------------------------
// qkvh: [B*L][768] f16 (q|k|v).  Produces:
//   qh,kh : [b*H+h][L][32] f16 (RoPE applied)
//   vt    : [b*H+h][32][L] f16 (transposed for PV B-operand)

__global__ void rope_pack_kernel(const _Float16* __restrict__ qkvh,
                                 _Float16* __restrict__ qh, _Float16* __restrict__ kh,
                                 _Float16* __restrict__ vt) {
  int tid = blockIdx.x * blockDim.x + threadIdx.x;  // over B*H*L
  if (tid >= B_DIM * H_DIM * L_DIM) return;
  int l = tid % L_DIM;
  int bh = tid / L_DIM;
  int h = bh % H_DIM;
  int b = bh / H_DIM;

  const _Float16* src = qkvh + (size_t)(b * L_DIM + l) * 768 + h * 32;
  _Float16* qdst = qh + ((size_t)bh * L_DIM + l) * 32;
  _Float16* kdst = kh + ((size_t)bh * L_DIM + l) * 32;
  _Float16* vdst = vt + (size_t)bh * 32 * L_DIM + l;

  const float lntheta = -0.5756462732485115f;  // -ln(10000)/16
#pragma unroll
  for (int j = 0; j < 16; ++j) {
    float theta = __expf((float)j * lntheta);
    float ang = (float)l * theta;
    float s, c;
    __sincosf(ang, &s, &c);
    float q0 = (float)src[2 * j],       q1 = (float)src[2 * j + 1];
    float k0 = (float)src[256 + 2 * j], k1 = (float)src[256 + 2 * j + 1];
    qdst[2 * j]     = (_Float16)(q0 * c - q1 * s);
    qdst[2 * j + 1] = (_Float16)(q1 * c + q0 * s);
    kdst[2 * j]     = (_Float16)(k0 * c - k1 * s);
    kdst[2 * j + 1] = (_Float16)(k1 * c + k0 * s);
    vdst[(size_t)(2 * j) * L_DIM]     = src[512 + 2 * j];
    vdst[(size_t)(2 * j + 1) * L_DIM] = src[512 + 2 * j + 1];
  }
}

// --------------------------- fused attention -------------------------------
// One workgroup (8 waves) per (bh, 16-row q-tile).
// LDS: slog[16][2048] f32 (128 KB) + spart[8][512] f32 (16 KB).
// Phase 1: logits -> LDS (each wave: 16 k-tiles, 1 WMMA each)
// Phase 2: softmax per row in LDS (wave32 shuffle reductions, 2 rows/wave)
// Phase 3: async LDS->global stream of the normalized series block
// Phase 4: PV, A-operand from LDS, K-split across waves, LDS partial reduce.

#define FUSED_LDS_BYTES (16 * 2048 * 4 + 8 * 512 * 4)

__global__ void fused_attn_kernel(const _Float16* __restrict__ qh,
                                  const _Float16* __restrict__ kh,
                                  const _Float16* __restrict__ vt,
                                  float* __restrict__ series,
                                  _Float16* __restrict__ resh) {
  extern __shared__ char smem[];
  float* slog  = (float*)smem;                       // [16][2048]
  float* spart = (float*)(smem + 16 * 2048 * 4);     // [8][512]

  int wave = threadIdx.x >> 5;
  int lane = threadIdx.x & 31;
  int bh = blockIdx.x >> 7;   // 0..31
  int qt = blockIdx.x & 127;  // 0..127

  int ko   = (lane & 16) ? 8 : 0;
  int mrow = (lane & 16) ? 8 : 0;
  int ncol = lane & 15;

  // ---- Phase 1: logits into LDS ----
  const _Float16* qrow = qh + ((size_t)bh * L_DIM + qt * 16 + (lane & 15)) * 32;
  v16h qa = ldtile_h(qrow, ko);
  const float scale = 0.17677669529663688f;  // 1/sqrt(32)

  const _Float16* kbase = kh + (size_t)bh * L_DIM * 32;
  for (int kt = wave; kt < 128; kt += 8) {
    const _Float16* krow = kbase + (size_t)(kt * 16 + (lane & 15)) * 32;
    __builtin_prefetch(krow + 8 * 16 * 32, 0, 0);  // next k-tile for this wave
    v16h kb = ldtile_h(krow, ko);
    v8f c = {};
    c = wmma_f16(qa, kb, c);
#pragma unroll
    for (int i = 0; i < 8; ++i)
      slog[(mrow + i) * 2048 + kt * 16 + ncol] = c[i] * scale;
  }
  __syncthreads();

  // ---- Phase 2: softmax (wave w owns rows 2w, 2w+1) ----
#pragma unroll
  for (int rr = 0; rr < 2; ++rr) {
    float* row = slog + (wave * 2 + rr) * 2048;
    float m = -3.402823466e+38f;
    for (int i = lane; i < 2048; i += 32) m = fmaxf(m, row[i]);
#pragma unroll
    for (int off = 16; off > 0; off >>= 1) m = fmaxf(m, __shfl_xor(m, off, 32));
    float sum = 0.0f;
    for (int i = lane; i < 2048; i += 32) {
      float e = __expf(row[i] - m);
      row[i] = e;
      sum += e;
    }
#pragma unroll
    for (int off = 16; off > 0; off >>= 1) sum += __shfl_xor(sum, off, 32);
    float inv = 1.0f / sum;
    for (int i = lane; i < 2048; i += 32) row[i] *= inv;
  }
  __syncthreads();

  // ---- Phase 3: stream series block LDS -> global via async store ----
  // 128 KB per workgroup; 16 B per lane per op; 32 ops per thread-slot.
  {
    float* gout = series + ((size_t)bh * L_DIM + qt * 16) * L_DIM;
    unsigned long long gbase = (unsigned long long)(uintptr_t)gout;
    unsigned ldsbase = (unsigned)(uintptr_t)slog;  // low 32 bits = LDS offset
#pragma unroll 4
    for (int it = 0; it < 32; ++it) {
      unsigned boff = (unsigned)(it * 256 + threadIdx.x) * 16u;
      unsigned long long ga = gbase + boff;
      unsigned la = ldsbase + boff;
      asm volatile("global_store_async_from_lds_b128 %0, %1, off"
                   :: "v"(ga), "v"(la)
                   : "memory");
    }
  }

  // ---- Phase 4: PV, K-split across waves (overlaps async drain) ----
  int k0 = wave * 256;
  const float* arow = slog + (lane & 15) * 2048 + k0;
  const _Float16* v0row = vt + ((size_t)bh * 32 + (lane & 15)) * L_DIM + k0;
  const _Float16* v1row = v0row + (size_t)16 * L_DIM;

  v8f acc0 = {}, acc1 = {};
#pragma unroll 2
  for (int kt = 0; kt < 256; kt += 32) {
    v16h a  = ldtile_f(arow + kt, ko);
    v16h b0 = ldtile_h(v0row + kt, ko);
    v16h b1 = ldtile_h(v1row + kt, ko);
    acc0 = wmma_f16(a, b0, acc0);
    acc1 = wmma_f16(a, b1, acc1);
  }

  float* p = spart + wave * 512;  // [16][32]
#pragma unroll
  for (int i = 0; i < 8; ++i) {
    p[(mrow + i) * 32 + ncol]      = acc0[i];
    p[(mrow + i) * 32 + 16 + ncol] = acc1[i];
  }

  // Drain async LDS->global stores before the final barrier/epilogue.
  asm volatile("s_wait_asynccnt 0" ::: "memory");
  __syncthreads();

  int h = bh % H_DIM;
  int b = bh / H_DIM;
  for (int o = threadIdx.x; o < 512; o += 256) {
    float s = 0.0f;
#pragma unroll
    for (int w2 = 0; w2 < 8; ++w2) s += spart[w2 * 512 + o];
    int r = o >> 5;
    int d = o & 31;
    resh[(size_t)(b * L_DIM + qt * 16 + r) * 256 + h * 32 + d] = (_Float16)s;
  }
}

// --------------------------- launch ----------------------------------------

extern "C" void kernel_launch(void* const* d_in, const int* in_sizes, int n_in,
                              void* d_out, int out_size, void* d_ws, size_t ws_size,
                              hipStream_t stream) {
  (void)in_sizes; (void)n_in; (void)out_size; (void)ws_size;

  const float* x  = (const float*)d_in[0];
  const float* Wq = (const float*)d_in[1];
  const float* bq = (const float*)d_in[2];
  const float* Wk = (const float*)d_in[3];
  const float* bk = (const float*)d_in[4];
  const float* Wv = (const float*)d_in[5];
  const float* bv = (const float*)d_in[6];
  const float* Wo = (const float*)d_in[7];
  const float* bo = (const float*)d_in[8];
  // d_in[9] = patch_index (-1): series passes through unchanged.

  float* out    = (float*)d_out;                                  // [B,L,256]
  float* series = (float*)d_out + (size_t)B_DIM * L_DIM * M_DIM;  // [B,H,L,L]

  // ---- workspace carve-up (bytes) ----
  char* ws = (char*)d_ws;
  size_t off = 0;
  _Float16* xh    = (_Float16*)(ws + off); off += (size_t)8192 * 256 * 2;     //  4 MB
  _Float16* qkvh  = (_Float16*)(ws + off); off += (size_t)8192 * 768 * 2;     // 12 MB
  _Float16* qh    = (_Float16*)(ws + off); off += (size_t)32 * 2048 * 32 * 2; //  4 MB
  _Float16* kh    = (_Float16*)(ws + off); off += (size_t)32 * 2048 * 32 * 2; //  4 MB
  _Float16* vt    = (_Float16*)(ws + off); off += (size_t)32 * 32 * 2048 * 2; //  4 MB
  _Float16* resh  = (_Float16*)(ws + off); off += (size_t)8192 * 256 * 2;     //  4 MB
  _Float16* Wt    = (_Float16*)(ws + off); off += (size_t)768 * 256 * 2;      // 384 KB
  _Float16* Wto   = (_Float16*)(ws + off); off += (size_t)256 * 256 * 2;      // 128 KB
  float*    bcat  = (float*)(ws + off);    off += 768 * 4;

  // 1) pack inputs to f16 / transposed layouts
  pack_x_kernel<<<8192, 256, 0, stream>>>(x, xh, 8192 * 256);
  pack_wqkv_kernel<<<768, 256, 0, stream>>>(Wq, Wk, Wv, bq, bk, bv, Wt, bcat);
  pack_wo_kernel<<<256, 256, 0, stream>>>(Wo, Wto);

  // 2) fused QKV projection: [8192,256] x [256,768] -> qkvh f16
  gemm_wmma_kernel<<<3072, 256, 0, stream>>>(xh, Wt, bcat, nullptr, qkvh,
                                             8192, 768, 256);

  // 3) RoPE + relayout (B*H*L = 65536 threads)
  rope_pack_kernel<<<256, 256, 0, stream>>>(qkvh, qh, kh, vt);

  // 4) fused logits + softmax + async series-write + PV:
  //    32 bh * 128 q-tiles = 4096 workgroups, 144 KB dynamic LDS each
  fused_attn_kernel<<<4096, 256, FUSED_LDS_BYTES, stream>>>(qh, kh, vt, series,
                                                            resh);

  // 5) out = res @ Wo + bo: 8192 tiles -> 1024 blocks
  gemm_wmma_kernel<<<1024, 256, 0, stream>>>(resh, Wto, bo, out, nullptr,
                                             8192, 256, 256);
}
